// GCNModelWithWeights_11063835754876
// MI455X (gfx1250) — matrix operand order repaired and checked by reading, not compiled, not run on hardware
//
#include <hip/hip_runtime.h>
#include <hip/hip_bf16.h>

typedef __attribute__((ext_vector_type(16))) __bf16 v16bf;
typedef __attribute__((ext_vector_type(8)))  __bf16 v8bf;
typedef __attribute__((ext_vector_type(8)))  float  v8f;

#define N_NODES 100000
#define N_EDGES 1600000
#define IN_DIM  128
#define HID_DIM 128
#define OUT_DIM 64

// float -> bf16, round-to-nearest-even via integer ops (used only in one-shot
// conversion / pack kernels, never in GEMM inner loops)
static __device__ __forceinline__ __bf16 f2bf(float f) {
  unsigned int u = __float_as_uint(f);
  unsigned int r = (u + 0x7FFFu + ((u >> 16) & 1u)) >> 16;
  unsigned short s = (unsigned short)r;
  return __builtin_bit_cast(__bf16, s);
}

// ---------------------------------------------------------------- utility ---
__global__ void fill_f32(float* __restrict__ p, long long n, float v) {
  long long i = (long long)blockIdx.x * blockDim.x + threadIdx.x;
  long long stride = (long long)gridDim.x * blockDim.x;
  for (; i < n; i += stride) p[i] = v;
}

// one-shot fp32 -> bf16 conversion; each thread handles 8 elements (16B store)
__global__ void convert_bf16(const float* __restrict__ src, __bf16* __restrict__ dst,
                             long long n8) {
  long long t = (long long)blockIdx.x * blockDim.x + threadIdx.x;
  if (t >= n8) return;
  const float4* p = (const float4*)src + t * 2;
  float4 f0 = p[0], f1 = p[1];
  v8bf o;
  o[0] = f2bf(f0.x); o[1] = f2bf(f0.y); o[2] = f2bf(f0.z); o[3] = f2bf(f0.w);
  o[4] = f2bf(f1.x); o[5] = f2bf(f1.y); o[6] = f2bf(f1.z); o[7] = f2bf(f1.w);
  ((v8bf*)dst)[t] = o;
}

__global__ void degree_kernel(const int* __restrict__ col,
                              const float* __restrict__ w,
                              float* __restrict__ deg) {
  int e = blockIdx.x * blockDim.x + threadIdx.x;
  if (e < N_EDGES) atomicAdd(&deg[col[e]], w[e]);
}

__global__ void dis_kernel(const float* __restrict__ deg, float* __restrict__ dis) {
  int i = blockIdx.x * blockDim.x + threadIdx.x;
  if (i < N_NODES) dis[i] = rsqrtf(deg[i]);  // deg >= 1 (self-loop folded into init)
}

// --------------------------------------------------------------- WMMA GEMM ---
// C[N,NOUT] = A[N,128](bf16) @ B[NOUT,128](bf16, row-major)^T, fp32 accumulate.
// One wave32 per 16x16 output tile; NOUT/16 waves per block cover one 16-row
// stripe. Pure b128 loads + register shuffles + v_wmma per k-step.
template <int NOUT>
__global__ void gemm_bf16_wmma(const __bf16* __restrict__ A, const __bf16* __restrict__ B,
                               float* __restrict__ C) {
  const int lane  = threadIdx.x & 31;
  const int wave  = threadIdx.x >> 5;          // ntile: 0..NOUT/16-1
  const int mtile = blockIdx.x;                // 0..6249
  const int half  = lane >> 4;                 // 0 or 1
  const int nrow  = (wave << 4) + (lane & 15);
  const __bf16* arow = A + (size_t)((mtile << 4) + (lane & 15)) * 128;
  const __bf16* brow = B + (size_t)nrow * 128;
  v8f acc = {};
#pragma unroll
  for (int kk = 0; kk < 128; kk += 32) {
    // A fragment: lane holds row m=lane&15, K runs [8h, 8h+8) and [16+8h, 16+8h+8)
    v8bf a0 = *(const v8bf*)(arow + kk + half * 8);
    v8bf a1 = *(const v8bf*)(arow + kk + 16 + half * 8);
    // B fragment: lane holds col n, K runs [8h, 8h+8) and [16+8h, 16+8h+8)
    v8bf b0 = *(const v8bf*)(brow + kk + half * 8);
    v8bf b1 = *(const v8bf*)(brow + kk + 16 + half * 8);
    v16bf a = __builtin_shufflevector(a0, a1, 0, 1, 2, 3, 4, 5, 6, 7,
                                      8, 9, 10, 11, 12, 13, 14, 15);
    v16bf b = __builtin_shufflevector(b0, b1, 0, 1, 2, 3, 4, 5, 6, 7,
                                      8, 9, 10, 11, 12, 13, 14, 15);
    acc = __builtin_amdgcn_wmma_f32_16x16x32_bf16(false, a, false, b,
                                                  (short)0, acc, false, false);
  }
#pragma unroll
  for (int r = 0; r < 8; ++r) {
    int m = (mtile << 4) + (half << 3) + r;  // D: acc[r] = C[r + 8*half][lane&15]
    C[(size_t)m * NOUT + nrow] = acc[r];
  }
}

// ----------------------------------------------------------- edge scatter ---
// one wave32 per edge; lane handles 4 features (128-wide)
__global__ void aggregate128(const int* __restrict__ row, const int* __restrict__ col,
                             const float* __restrict__ w, const float* __restrict__ dis,
                             const float* __restrict__ h, float* __restrict__ agg) {
  int wv = blockIdx.x * (blockDim.x >> 5) + (threadIdx.x >> 5);
  if (wv >= N_EDGES) return;
  int lane = threadIdx.x & 31;
  int r = row[wv], c = col[wv];
  float norm = dis[r] * w[wv] * dis[c];
  float4 v = ((const float4*)(h + (size_t)r * 128))[lane];
  float* dst = agg + (size_t)c * 128 + lane * 4;
  atomicAdd(dst + 0, v.x * norm);
  atomicAdd(dst + 1, v.y * norm);
  atomicAdd(dst + 2, v.z * norm);
  atomicAdd(dst + 3, v.w * norm);
}

// one wave32 per edge; lane handles 2 features (64-wide)
__global__ void aggregate64(const int* __restrict__ row, const int* __restrict__ col,
                            const float* __restrict__ w, const float* __restrict__ dis,
                            const float* __restrict__ h, float* __restrict__ agg) {
  int wv = blockIdx.x * (blockDim.x >> 5) + (threadIdx.x >> 5);
  if (wv >= N_EDGES) return;
  int lane = threadIdx.x & 31;
  int r = row[wv], c = col[wv];
  float norm = dis[r] * w[wv] * dis[c];
  float2 v = ((const float2*)(h + (size_t)r * 64))[lane];
  float* dst = agg + (size_t)c * 64 + lane * 2;
  atomicAdd(dst + 0, v.x * norm);
  atomicAdd(dst + 1, v.y * norm);
}

// ------------------------------------------------------------ bias + ReLU ---
// layer1: out = relu(agg + h*dis^2 + b1) packed to bf16 (feeds GEMM2 directly)
__global__ void bias_relu1(const float* __restrict__ h, const float* __restrict__ agg,
                           const float* __restrict__ dis, const float* __restrict__ b,
                           __bf16* __restrict__ out) {
  long long idx = (long long)blockIdx.x * blockDim.x + threadIdx.x;
  if (idx >= (long long)N_NODES * HID_DIM) return;
  int i = (int)(idx >> 7), f = (int)(idx & 127);
  float d = dis[i];
  float v = agg[idx] + h[idx] * d * d + b[f];
  out[idx] = f2bf(fmaxf(v, 0.0f));
}

// layer2: out = relu(agg + h*dis^2 + b2) kept fp32
__global__ void bias_relu2(const float* __restrict__ h, const float* __restrict__ agg,
                           const float* __restrict__ dis, const float* __restrict__ b,
                           float* __restrict__ out) {
  long long idx = (long long)blockIdx.x * blockDim.x + threadIdx.x;
  if (idx >= (long long)N_NODES * OUT_DIM) return;
  int i = (int)(idx >> 6), f = (int)(idx & 63);
  float d = dis[i];
  float v = agg[idx] + h[idx] * d * d + b[f];
  out[idx] = fmaxf(v, 0.0f);
}

// ------------------------------------------------------- column softmax ----
// per-column max over 100k rows; post-ReLU values >= 0 so int-bits atomicMax is valid
__global__ void colmax_kernel(const float* __restrict__ fin, float* __restrict__ colmax) {
  __shared__ float sm[256];
  int t = threadIdx.x;
  int colf = t & 63, rg = t >> 6;
  int base = blockIdx.x * 256;
  float m = 0.0f;
  for (int r = rg; r < 256; r += 4) {
    int rowi = base + r;
    if (rowi < N_NODES) m = fmaxf(m, fin[(size_t)rowi * OUT_DIM + colf]);
  }
  sm[t] = m;
  __syncthreads();
  if (t < 64) {
    float mm = fmaxf(fmaxf(sm[t], sm[t + 64]), fmaxf(sm[t + 128], sm[t + 192]));
    atomicMax((int*)&colmax[t], __float_as_int(mm));
  }
}

__global__ void colsum_kernel(const float* __restrict__ fin, const float* __restrict__ colmax,
                              float* __restrict__ colsum) {
  __shared__ float sm[256];
  int t = threadIdx.x;
  int colf = t & 63, rg = t >> 6;
  int base = blockIdx.x * 256;
  float mx = colmax[colf];
  float s = 0.0f;
  for (int r = rg; r < 256; r += 4) {
    int rowi = base + r;
    if (rowi < N_NODES) s += expf(fin[(size_t)rowi * OUT_DIM + colf] - mx);
  }
  sm[t] = s;
  __syncthreads();
  if (t < 64) {
    float ss = sm[t] + sm[t + 64] + sm[t + 128] + sm[t + 192];
    atomicAdd(&colsum[t], ss);
  }
}

// out is the transposed result: out[f*N + i]
__global__ void softmax_write(const float* __restrict__ fin, const float* __restrict__ colmax,
                              const float* __restrict__ colsum, float* __restrict__ out) {
  long long idx = (long long)blockIdx.x * blockDim.x + threadIdx.x;
  if (idx >= (long long)OUT_DIM * N_NODES) return;
  int f = (int)(idx / N_NODES);
  int i = (int)(idx - (long long)f * N_NODES);
  out[idx] = expf(fin[(size_t)i * OUT_DIM + f] - colmax[f]) / colsum[f];
}

// ------------------------------------------------------------------ launch ---
extern "C" void kernel_launch(void* const* d_in, const int* in_sizes, int n_in,
                              void* d_out, int out_size, void* d_ws, size_t ws_size,
                              hipStream_t stream) {
  (void)in_sizes; (void)n_in; (void)out_size; (void)ws_size;
  const float* x  = (const float*)d_in[0];
  const int*   ei = (const int*)d_in[1];       // [2, E]
  const float* ew = (const float*)d_in[2];
  const float* W1 = (const float*)d_in[3];
  const float* b1 = (const float*)d_in[4];
  const float* W2 = (const float*)d_in[5];
  const float* b2 = (const float*)d_in[6];
  float* out = (float*)d_out;

  const size_t N = N_NODES;
  float* ws   = (float*)d_ws;
  float* deg  = ws;                       // [N]
  float* dis  = ws + N;                   // [N]
  float* bufA = ws + 2 * N;               // h1 [N*128] f32, later agg2 [N*64] f32
  float* bufB = ws + 2 * N + 128 * N;     // agg1 [N*128] f32, later h2 [N*64] f32
  float* bufC = ws + 2 * N + 256 * N;     // h1b [N*128] bf16, later final [N*64] f32
  float* bufX = ws + 2 * N + 320 * N;     // Xbf [N*128] bf16 (= N*64 float slots)
  float* bufW = ws + 2 * N + 384 * N;     // W1bf (8192 slots) + W2bf (4096 slots)
  float* colmax = bufW + 8192 + 4096;     // [64]
  float* colsum = colmax + 64;            // [64]
  __bf16* h1b  = (__bf16*)bufC;
  __bf16* Xbf  = (__bf16*)bufX;
  __bf16* W1bf = (__bf16*)bufW;
  __bf16* W2bf = (__bf16*)(bufW + 8192);

  const int* rowPtr = ei;                 // edge_index[0]
  const int* colPtr = ei + N_EDGES;       // edge_index[1]

  // init: deg=1 (self-loop weight), agg1=0, colmax/colsum=0
  fill_f32<<<1024, 256, 0, stream>>>(deg, (long long)N, 1.0f);
  fill_f32<<<4096, 256, 0, stream>>>(bufB, (long long)N * 128, 0.0f);
  fill_f32<<<1, 128, 0, stream>>>(colmax, 128, 0.0f);

  // one-shot bf16 mirrors of X, W1, W2
  convert_bf16<<<(int)((N * 16 + 255) / 256), 256, 0, stream>>>(x, Xbf, (long long)N * 16);
  convert_bf16<<<8, 256, 0, stream>>>(W1, W1bf, (HID_DIM * IN_DIM) / 8);
  convert_bf16<<<4, 256, 0, stream>>>(W2, W2bf, (OUT_DIM * HID_DIM) / 8);

  degree_kernel<<<(N_EDGES + 255) / 256, 256, 0, stream>>>(colPtr, ew, deg);
  dis_kernel<<<(N_NODES + 255) / 256, 256, 0, stream>>>(deg, dis);

  // layer 1
  gemm_bf16_wmma<128><<<N_NODES / 16, 256, 0, stream>>>(Xbf, W1bf, bufA);
  aggregate128<<<N_EDGES / 8, 256, 0, stream>>>(rowPtr, colPtr, ew, dis, bufA, bufB);
  bias_relu1<<<(int)((N * 128 + 255) / 256), 256, 0, stream>>>(bufA, bufB, dis, b1, h1b);

  // layer 2 (agg2 reuses bufA, h2 reuses bufB; safe: stream-ordered after last reads)
  fill_f32<<<4096, 256, 0, stream>>>(bufA, (long long)N * 64, 0.0f);
  gemm_bf16_wmma<64><<<N_NODES / 16, 128, 0, stream>>>(h1b, W2bf, bufB);
  aggregate64<<<N_EDGES / 8, 256, 0, stream>>>(rowPtr, colPtr, ew, dis, bufB, bufA);
  bias_relu2<<<(int)((N * 64 + 255) / 256), 256, 0, stream>>>(bufB, bufA, dis, b2, bufC);

  // softmax over nodes per output dim, then transposed write
  colmax_kernel<<<(N_NODES + 255) / 256, 256, 0, stream>>>(bufC, colmax);
  colsum_kernel<<<(N_NODES + 255) / 256, 256, 0, stream>>>(bufC, colmax, colsum);
  softmax_write<<<(int)(((long long)OUT_DIM * N + 255) / 256), 256, 0, stream>>>(
      bufC, colmax, colsum, out);
}
